// Encoder_25537875542226
// MI455X (gfx1250) — 8-bit matrix operand order repaired and checked by reading, compiled, not run on hardware
//
#include <hip/hip_runtime.h>
#include <hip/hip_bf16.h>
#include <stdint.h>


typedef int v8i __attribute__((ext_vector_type(8)));

#define BATCH      512
#define NPIX       784
#define DIMS       1100
#define NLEV       256
#define NCLASS     256
#define KCHUNKS    13

__device__ __forceinline__ int class_ndims(int c) { return (c >= 180) ? 5 : 4; }
__device__ __forceinline__ int class_dim(int c, int j) {
    if (c >= 180) { if (j == 0) return c - 180; --j; }
    return 76 + c + 256 * j;
}

__global__ void k_quant(const float* __restrict__ x, uint8_t* __restrict__ q, int n) {
    int i = blockIdx.x * blockDim.x + threadIdx.x;
    if (i >= n) return;
    float v = x[i] * (1.0f / 256.0f);
    v = fminf(fmaxf(v, 0.0f), 1.0f);
    q[i] = (uint8_t)(int)rintf(v * 255.0f);
}

__global__ void k_packP(const float* __restrict__ p, int8_t* __restrict__ pq, int n) {
    int i = blockIdx.x * blockDim.x + threadIdx.x;
    if (i >= n) return;
    pq[i] = (p[i] > 0.0f) ? (int8_t)1 : (int8_t)-1;
}

__global__ void k_thresh(const float* __restrict__ V, uint8_t* __restrict__ T) {
    int d = blockIdx.x * blockDim.x + threadIdx.x;
    if (d >= DIMS) return;
    int t = 0;
    for (int k = 0; k < NLEV; ++k) t += (V[k * DIMS + d] < 0.0f) ? 1 : 0;
    T[d] = (uint8_t)t;
}

__global__ void k_colsum(const int8_t* __restrict__ pq, int* __restrict__ S) {
    int d = blockIdx.x * blockDim.x + threadIdx.x;
    if (d >= DIMS) return;
    int s = 0;
    for (int n = 0; n < NPIX; ++n) s += (int)pq[n * DIMS + d];
    S[d] = s;
}

__global__ void k_bfrag(const int8_t* __restrict__ pq, int* __restrict__ bfrag) {
    int kc = blockIdx.x;
    int c  = blockIdx.y;
    int tid = threadIdx.x;
    int lane = tid >> 3;
    int v    = tid & 7;
    int j     = lane & 15;
    int khalf = lane >> 4;
    int nd = class_ndims(c);
    int kbase = (v >> 1) * 16 + khalf * 8 + (v & 1) * 4;
    unsigned int dw = 0;
    if (j < nd) {
        int d = class_dim(c, j);
        #pragma unroll
        for (int by = 0; by < 4; ++by) {
            int k = kc * 64 + kbase + by;
            int8_t val = (k < NPIX) ? pq[k * DIMS + d] : (int8_t)0;
            dw |= ((unsigned int)(uint8_t)val) << (8 * by);
        }
    }
    bfrag[(((c * KCHUNKS) + kc) * 32 + lane) * 8 + v] = (int)dw;
}

__device__ __forceinline__ unsigned int ge01(unsigned int x, unsigned int ny,
                                             unsigned int K) {
    const unsigned int NH = 0x7f7f7f7fu, H = 0x80808080u;
    unsigned int p  = (x & NH) + K;
    unsigned int ge = ((x & ny) | ((x | ny) & p)) & H;
    return ge >> 7;
}

__global__ __launch_bounds__(32) void k_gemm(const uint8_t* __restrict__ Q,
                                             const uint8_t* __restrict__ T,
                                             const int*     __restrict__ S,
                                             const int*     __restrict__ bfrag,
                                             float*         __restrict__ out) {
    int lane = threadIdx.x;
    int b0   = blockIdx.x * 16;
    int c0   = blockIdx.y * 2;
    int c1   = c0 + 1;

    unsigned int t0 = T[class_dim(c0, 0)];
    unsigned int t1 = T[class_dim(c1, 0)];
    unsigned int ny0 = ~(t0 * 0x01010101u), K0 = ((((~t0) & 0x7fu) + 1u) * 0x01010101u);
    unsigned int ny1 = ~(t1 * 0x01010101u), K1 = ((((~t1) & 0x7fu) + 1u) * 0x01010101u);

    int M     = lane & 15;
    int khalf = lane >> 4;
    const uint8_t* qrow  = Q + (size_t)(b0 + M) * NPIX;
    const int* bb0 = bfrag + ((size_t)c0 * KCHUNKS * 32 + lane) * 8;
    const int* bb1 = bfrag + ((size_t)c1 * KCHUNKS * 32 + lane) * 8;

    v8i acc0 = {0, 0, 0, 0, 0, 0, 0, 0};
    v8i acc1 = {0, 0, 0, 0, 0, 0, 0, 0};

    #define DO_CLASS(Q8, NY, KK, ACC, BP)                                         \
        {                                                                          \
            v8i a;                                                                 \
            _Pragma("unroll")                                                      \
            for (int g = 0; g < 4; ++g) {                                          \
                a[2 * g]     = (int)ge01((unsigned int)(Q8[g]),         NY, KK);   \
                a[2 * g + 1] = (int)ge01((unsigned int)(Q8[g] >> 32),   NY, KK);   \
            }                                                                      \
            v8i bmat = *(const v8i*)(BP);                                          \
            ACC = __builtin_amdgcn_wmma_i32_16x16x64_iu8(                          \
                 false, a,  true, bmat, ACC, false, false);      \
        }

    for (int kc = 0; kc < KCHUNKS - 1; ++kc) {
        unsigned long long q8[4];
        #pragma unroll
        for (int g = 0; g < 4; ++g)
            q8[g] = *(const unsigned long long*)(qrow + kc * 64 + g * 16 + khalf * 8);
        DO_CLASS(q8, ny0, K0, acc0, bb0 + kc * 32 * 8)
        DO_CLASS(q8, ny1, K1, acc1, bb1 + kc * 32 * 8)
    }
    {
        unsigned long long q8[4] = {0ull, 0ull, 0ull, 0ull};
        q8[0] = *(const unsigned long long*)(qrow + 12 * 64 + khalf * 8);
        DO_CLASS(q8, ny0, K0, acc0, bb0 + 12 * 32 * 8)
        DO_CLASS(q8, ny1, K1, acc1, bb1 + 12 * 32 * 8)
    }
    #undef DO_CLASS

    int j = lane & 15;
    #pragma unroll
    for (int pass = 0; pass < 2; ++pass) {
        int c  = pass ? c1 : c0;
        v8i ac = pass ? acc1 : acc0;
        if (j < class_ndims(c)) {
            int d  = class_dim(c, j);
            int hs = S[d] / 2;
            #pragma unroll
            for (int r = 0; r < 8; ++r) {
                int row = b0 + r + 8 * khalf;
                out[(size_t)row * DIMS + d] = (ac[r] > hs) ? 1.0f : -1.0f;
            }
        }
    }
}

extern "C" void kernel_launch(void* const* d_in, const int* in_sizes, int n_in,
                              void* d_out, int out_size, void* d_ws, size_t ws_size,
                              hipStream_t stream) {
    (void)in_sizes; (void)n_in; (void)out_size; (void)ws_size;

    const float* x = (const float*)d_in[0];
    const float* P = (const float*)d_in[1];
    const float* V = (const float*)d_in[2];
    float* out = (float*)d_out;

    char* ws = (char*)d_ws;
    uint8_t* Q     = (uint8_t*)(ws);
    int8_t*  Pq    = (int8_t*) (ws + 401408);
    uint8_t* T     = (uint8_t*)(ws + 1263808);
    int*     S     = (int*)    (ws + 1264928);
    int*     Bfrag = (int*)    (ws + 1269344);

    k_quant <<<(BATCH * NPIX + 255) / 256, 256, 0, stream>>>(x, Q, BATCH * NPIX);
    k_packP <<<(NPIX * DIMS + 255) / 256, 256, 0, stream>>>(P, Pq, NPIX * DIMS);
    k_thresh<<<(DIMS + 255) / 256,        256, 0, stream>>>(V, T);
    k_colsum<<<(DIMS + 255) / 256,        256, 0, stream>>>(Pq, S);
    k_bfrag <<<dim3(KCHUNKS, NCLASS),     256, 0, stream>>>(Pq, Bfrag);
    k_gemm  <<<dim3(BATCH / 16, NCLASS / 2), 32, 0, stream>>>(Q, T, S, Bfrag, out);
}
